// DTWLoss_10144712753762
// MI455X (gfx1250) — matrix operand order repaired and checked by reading, not compiled, not run on hardware
//
#include <hip/hip_runtime.h>
#include <hip/hip_bf16.h>

// DTW loss for MI455X (gfx1250, wave32).
// Phase 1: row norms.
// Phase 2: cost = ||a||^2 + ||b||^2 - 2 a.b via f16 WMMA with hi/lo split
//          (a_hi*b_hi + a_hi*b_lo + a_lo*b_hi, f32 accumulate -> ~fp32
//          accuracy), written in SKEWED layout: cell (i,j) -> [(i+j)%L][i]
//          so each DTW anti-diagonal is a contiguous row segment.
// Phase 3: anti-diagonal DP, one workgroup per batch, coalesced cost reads
//          software-pipelined across the per-diagonal barrier.
// Phase 4: mean over batch.

typedef __attribute__((ext_vector_type(16))) _Float16 v16h;
typedef __attribute__((ext_vector_type(8)))  float    v8f;

#define BB 128
#define LL 384
#define DD 128
#define NT 24   // LL/16 tiles per dimension

// ---------------------------------------------------------------- row norms
__global__ void dtw_row_norms(const float* __restrict__ s1,
                              const float* __restrict__ s2,
                              float* __restrict__ n1,
                              float* __restrict__ n2) {
    int idx = blockIdx.x * blockDim.x + threadIdx.x;   // 0 .. 2*B*L-1
    const int R1 = BB * LL;
    const float* src;
    float* dst;
    int r;
    if (idx < R1) { src = s1; dst = n1; r = idx; }
    else          { src = s2; dst = n2; r = idx - R1; }
    const float4* p = (const float4*)(src + (size_t)r * DD);
    float acc = 0.0f;
#pragma unroll
    for (int q = 0; q < DD / 4; ++q) {
        float4 v = p[q];
        acc += v.x * v.x + v.y * v.y + v.z * v.z + v.w * v.w;
    }
    dst[r] = acc;
}

// ------------------------------------------------------------- cost (WMMA)
// One wave per 16x16 output tile. A = s1 tile rows (16xK), B = s2 tile rows
// loaded column-major (Kx16). f32 inputs split into f16 hi + f16 residual;
// dot accumulated as hi*hi + hi*lo + lo*hi in one f32 accumulator.
__global__ void dtw_cost_wmma(const float* __restrict__ s1,
                              const float* __restrict__ s2,
                              const float* __restrict__ n1,
                              const float* __restrict__ n2,
                              float* __restrict__ cost) {
    int wave = (blockIdx.x * blockDim.x + threadIdx.x) >> 5;
    int lane = threadIdx.x & 31;

    int b   = wave / (NT * NT);
    int rem = wave % (NT * NT);
    int ti  = rem / NT;
    int tj  = rem % NT;
    if (b >= BB) return;

    int r16 = lane & 15;   // row within tile (A: M, B: N)
    int hg  = lane >> 4;   // lane half-group

    const float* arow = s1 + (((size_t)b * LL) + (size_t)(ti * 16 + r16)) * DD;
    const float* brow = s2 + (((size_t)b * LL) + (size_t)(tj * 16 + r16)) * DD;

    const int abase = hg * 16;   // A contiguous:    K = abase + h
    const int bbase = hg * 16;   // B col-major:     K = bbase + h

    v8f acc = {};
#pragma unroll
    for (int k0 = 0; k0 < DD; k0 += 32) {
        v16h ah, al, bh, bl;
#pragma unroll
        for (int h = 0; h < 16; ++h) {
            float av = arow[k0 + abase + h];
            float bv = brow[k0 + bbase + h];
            _Float16 avh = (_Float16)av;
            _Float16 bvh = (_Float16)bv;
            ah[h] = avh;
            bh[h] = bvh;
            al[h] = (_Float16)(av - (float)avh);
            bl[h] = (_Float16)(bv - (float)bvh);
        }
        // dot ~= ah*bh + ah*bl + al*bh  (lo*lo term below f32 epsilon)
        acc = __builtin_amdgcn_wmma_f32_16x16x32_f16(
            false, ah, false, bh, (short)0, acc, false, false);
        acc = __builtin_amdgcn_wmma_f32_16x16x32_f16(
            false, ah, false, bl, (short)0, acc, false, false);
        acc = __builtin_amdgcn_wmma_f32_16x16x32_f16(
            false, al, false, bh, (short)0, acc, false, false);
    }

    // C/D layout: lane holds N = lane&15, rows M = r + 8*(lane>>4), r = 0..7
    int   n   = r16;
    int   mOf = hg * 8;
    int   j   = tj * 16 + n;
    float nb  = n2[(size_t)b * LL + j];
    const float* n1b = n1 + (size_t)b * LL + ti * 16;
    float* cb = cost + (size_t)b * LL * LL;
#pragma unroll
    for (int r = 0; r < 8; ++r) {
        int m = mOf + r;
        int i = ti * 16 + m;
        float c = n1b[m] + nb - 2.0f * acc[r];
        int rs = (i + j) % LL;              // skewed row = anti-diagonal id mod L
        cb[(size_t)rs * LL + i] = c;        // column = i  (bijective per row)
    }
}

// ------------------------------------------------------------ DTW dynamic
// One block of 384 threads per batch. Anti-diagonal wavefront, 3 rotating
// diagonals in LDS. Cost reads are contiguous per diagonal (skewed layout)
// and pipelined one diagonal ahead in registers, prefetched two ahead.
__global__ void dtw_dp(const float* __restrict__ cost,
                       float* __restrict__ finals) {
    const int L = LL;
    int b = blockIdx.x;
    int t = threadIdx.x;
    __shared__ float diag[3][LL];

    const float* cb = cost + (size_t)b * L * L;
    const float INF = 3.0e38f;

    if (t == 0) diag[0][0] = cb[0];        // cell (0,0): skew row 0, col 0

    // Preload cost for diagonal d = 1 (cells (0,1),(1,0) -> skew row 1, cols 0,1)
    float cnext = (t <= 1) ? cb[(size_t)1 * L + t] : 0.0f;
    __syncthreads();

    for (int d = 1; d <= 2 * L - 2; ++d) {
        int lo = (d > L - 1) ? d - (L - 1) : 0;
        int hi = (d < L - 1) ? d : L - 1;
        int i  = lo + t;
        float ccur = cnext;

        // Issue next diagonal's cost load before touching LDS / barrier.
        if (d < 2 * L - 2) {
            int dn  = d + 1;
            int lon = (dn > L - 1) ? dn - (L - 1) : 0;
            int hin = (dn < L - 1) ? dn : L - 1;
            int in_ = lon + t;
            cnext = (in_ <= hin) ? cb[(size_t)(dn % L) * L + in_] : 0.0f;
        }
        if (d + 2 <= 2 * L - 2 && t == 0) {
            __builtin_prefetch(&cb[(size_t)((d + 2) % L) * L], 0, 0);
        }

        int pc = d % 3;
        int p1 = (d + 2) % 3;   // == (d-1) % 3
        int p2 = (d + 1) % 3;   // == (d-2) % 3
        if (i <= hi) {
            int j = d - i;
            float best = INF;
            if (i >= 1)           best = fminf(best, diag[p1][i - 1]); // DTW[i-1][j]
            if (j >= 1)           best = fminf(best, diag[p1][i]);     // DTW[i][j-1]
            if (i >= 1 && j >= 1) best = fminf(best, diag[p2][i - 1]); // DTW[i-1][j-1]
            diag[pc][i] = ccur + best;
        }
        __syncthreads();
    }

    if (t == 0) finals[b] = diag[(2 * L - 2) % 3][L - 1];
}

// --------------------------------------------------------------- reduction
__global__ void dtw_reduce(const float* __restrict__ finals,
                           float* __restrict__ out) {
    __shared__ float s[BB];
    int t = threadIdx.x;
    s[t] = finals[t];
    __syncthreads();
    for (int off = BB / 2; off > 0; off >>= 1) {
        if (t < off) s[t] += s[t + off];
        __syncthreads();
    }
    if (t == 0) out[0] = s[0] * (1.0f / (float)BB);
}

// ------------------------------------------------------------------ launch
extern "C" void kernel_launch(void* const* d_in, const int* in_sizes, int n_in,
                              void* d_out, int out_size, void* d_ws, size_t ws_size,
                              hipStream_t stream) {
    const float* s1 = (const float*)d_in[0];   // [B, L1, D] f32
    const float* s2 = (const float*)d_in[1];   // [B, L2, D] f32
    float* out = (float*)d_out;                // scalar

    // Workspace layout (floats): cost(skewed) | n1 | n2 | finals (~75.9 MB)
    float* cost   = (float*)d_ws;
    float* n1     = cost + (size_t)BB * LL * LL;
    float* n2     = n1 + (size_t)BB * LL;
    float* finals = n2 + (size_t)BB * LL;

    // Phase 1: row norms. 2*B*L rows, 256 threads/block.
    {
        int total  = 2 * BB * LL;
        int blocks = (total + 255) / 256;
        dtw_row_norms<<<blocks, 256, 0, stream>>>(s1, s2, n1, n2);
    }

    // Phase 2: cost matrix via WMMA (hi/lo split). One wave per 16x16 tile.
    {
        int tiles  = BB * NT * NT;          // 73728 tiles
        int blocks = tiles / 8;             // 8 waves (256 thr) per block
        dtw_cost_wmma<<<blocks, 256, 0, stream>>>(s1, s2, n1, n2, cost);
    }

    // Phase 3: DTW DP, one workgroup per batch.
    dtw_dp<<<BB, LL, 0, stream>>>(cost, finals);

    // Phase 4: batch mean.
    dtw_reduce<<<1, BB, 0, stream>>>(finals, out);
}
